// ChunkedCrossAttention_32744830665570
// MI455X (gfx1250) — matrix operand order repaired and checked by reading, not compiled, not run on hardware
//
#include <hip/hip_runtime.h>
#include <hip/hip_bf16.h>

// ---------------------------------------------------------------------------
// ChunkedCrossAttention for MI455X (gfx1250, wave32, WMMA, async-to-LDS).
//  - one upfront fp32->bf16 conversion pass (query, key, 4 weight matrices)
//  - pure-bf16 double-buffered GEMMs using GLOBAL_LOAD_ASYNC_TO_LDS_B128
//    (ASYNCcnt) to stage tiles while v_wmma_f32_16x16x32_bf16 consumes them
//  - fused per-(chunk,head) attention with WMMA for QK^T and P*V,
//    softmax row reductions via native LDS float atomics (ds_max_num_f32)
// ---------------------------------------------------------------------------

typedef __attribute__((ext_vector_type(16))) __bf16          v16bf;
typedef __attribute__((ext_vector_type(8)))  float           v8f;
typedef __attribute__((ext_vector_type(8)))  unsigned short  u16x8;
typedef __attribute__((ext_vector_type(4)))  int             v4i;

union BF16Frag {
    v16bf bf;
    u16x8 h[2];
};

__device__ inline unsigned short f32_to_bf16(float f) {
    unsigned int u = __float_as_uint(f);
    unsigned int r = u + 0x7FFFu + ((u >> 16) & 1u);   // round-to-nearest-even
    return (unsigned short)(r >> 16);
}

__device__ inline u16x8 cvt8(float4 a, float4 b) {
    u16x8 o;
    o[0] = f32_to_bf16(a.x); o[1] = f32_to_bf16(a.y);
    o[2] = f32_to_bf16(a.z); o[3] = f32_to_bf16(a.w);
    o[4] = f32_to_bf16(b.x); o[5] = f32_to_bf16(b.y);
    o[6] = f32_to_bf16(b.z); o[7] = f32_to_bf16(b.w);
    return o;
}

__device__ inline void atomicMaxF(float* a, float v) {
#if __has_builtin(__hip_atomic_fetch_max)
    __hip_atomic_fetch_max(a, v, __ATOMIC_RELAXED, __HIP_MEMORY_SCOPE_WORKGROUP);
#else
    int* ai = (int*)a;
    int cur = *ai;
    while (__int_as_float(cur) < v) {
        int old = atomicCAS(ai, cur, __float_as_int(v));
        if (old == cur) break;
        cur = old;
    }
#endif
}

// ---- CDNA5 async global->LDS copy (16 bytes per lane, ASYNCcnt-tracked) ----
__device__ inline void async_copy_b128(const void* gsrc, void* ldst) {
#if __has_builtin(__builtin_amdgcn_global_load_async_to_lds_b128)
    __builtin_amdgcn_global_load_async_to_lds_b128(
        (__attribute__((address_space(1))) v4i*)gsrc,
        (__attribute__((address_space(3))) v4i*)ldst,
        0, 0);
#else
    const unsigned           lo = (unsigned)(size_t)ldst;          // LDS offset
    const unsigned long long ga = (unsigned long long)(size_t)gsrc;
    asm volatile("global_load_async_to_lds_b128 %0, %1, off"
                 :: "v"(lo), "v"(ga) : "memory");
#endif
}

__device__ inline void wait_async0() {
#if __has_builtin(__builtin_amdgcn_s_wait_asynccnt)
    __builtin_amdgcn_s_wait_asynccnt(0);
#else
    asm volatile("s_wait_asynccnt 0x0" ::: "memory");
#endif
}

// ---------------------------------------------------------------------------
// fp32 -> bf16 elementwise conversion, 8 elems/thread (n % 2048 == 0 here)
// ---------------------------------------------------------------------------
__global__ __launch_bounds__(256) void cvt_kernel(
    const float* __restrict__ in, unsigned short* __restrict__ out, int n)
{
    const int i = (blockIdx.x * 256 + threadIdx.x) * 8;
    if (i >= n) return;
    const float4* s = (const float4*)(in + i);
    float4 a = s[0], b = s[1];
    *(u16x8*)(out + i) = cvt8(a, b);
}

// ---------------------------------------------------------------------------
// Pure-bf16 tiled GEMM: C[M,N] = A[M,K]*B[K,N] + bias[N]
// Block 256 threads (8 waves), tile 128x128, BK=32, double-buffered LDS
// staged via async-to-LDS. Waves 2(m) x 4(n); wave tile 64x32 = 4x2 WMMAs.
// Requires M%128==0, N%128==0, K%32==0.
// ---------------------------------------------------------------------------
template <typename OT>
__global__ __launch_bounds__(256) void gemm_bf16_kernel(
    const unsigned short* __restrict__ A, const unsigned short* __restrict__ B,
    const float* __restrict__ bias, OT* __restrict__ C,
    int M, int N, int K)
{
    __shared__ unsigned short sA[2][128 * 40];   // [row][k], stride 40 (pad 8)
    __shared__ unsigned short sB[2][32 * 136];   // [k][n],  stride 136 (pad 8)

    const int tid  = threadIdx.x;
    const int lane = tid & 31;
    const int w    = tid >> 5;
    const int wm   = w >> 2;          // 0..1
    const int wn   = w & 3;           // 0..3
    const int m0   = blockIdx.y * 128;
    const int n0   = blockIdx.x * 128;

    // staging coordinates for this thread (2 A-chunks + 2 B-chunks of 16B)
    const int ar0 = tid >> 2,         ac0 = (tid & 3) * 8;           // chunk tid
    const int ar1 = (tid + 256) >> 2, ac1 = ((tid + 256) & 3) * 8;   // chunk tid+256
    const int br0 = tid >> 4,         bc0 = (tid & 15) * 8;
    const int br1 = (tid + 256) >> 4, bc1 = ((tid + 256) & 15) * 8;

    v8f acc[4][2] = {};

    const int nk = K / 32;

    // prefetch first tiles into buffer 0
    {
        const int k0 = 0;
        async_copy_b128(A + (size_t)(m0 + ar0) * K + k0 + ac0, &sA[0][ar0 * 40 + ac0]);
        async_copy_b128(A + (size_t)(m0 + ar1) * K + k0 + ac1, &sA[0][ar1 * 40 + ac1]);
        async_copy_b128(B + (size_t)(k0 + br0) * N + n0 + bc0, &sB[0][br0 * 136 + bc0]);
        async_copy_b128(B + (size_t)(k0 + br1) * N + n0 + bc1, &sB[0][br1 * 136 + bc1]);
    }
    wait_async0();
    __syncthreads();

    int cur = 0;
    for (int ki = 0; ki < nk; ++ki) {
        // issue async stage of next tile into the other buffer
        if (ki + 1 < nk) {
            const int k0 = (ki + 1) * 32;
            const int nb = cur ^ 1;
            async_copy_b128(A + (size_t)(m0 + ar0) * K + k0 + ac0, &sA[nb][ar0 * 40 + ac0]);
            async_copy_b128(A + (size_t)(m0 + ar1) * K + k0 + ac1, &sA[nb][ar1 * 40 + ac1]);
            async_copy_b128(B + (size_t)(k0 + br0) * N + n0 + bc0, &sB[nb][br0 * 136 + bc0]);
            async_copy_b128(B + (size_t)(k0 + br1) * N + n0 + bc1, &sB[nb][br1 * 136 + bc1]);
        }

        // fragments + WMMA from the current buffer
        BF16Frag af[4], bf[2];
        const int m  = lane & 15;
        const int kb = (lane >> 4) * 8;   // lane half selects K base 0 / 8
#pragma unroll
        for (int mi = 0; mi < 4; ++mi) {
            const unsigned short* p = &sA[cur][(wm * 64 + mi * 16 + m) * 40 + kb];
            af[mi].h[0] = *(const u16x8*)p;          // K = kb .. kb+7
            af[mi].h[1] = *(const u16x8*)(p + 16);   // K = kb+16 .. kb+23
        }
#pragma unroll
        for (int ni = 0; ni < 2; ++ni) {
            const unsigned short* p = &sB[cur][kb * 136 + wn * 32 + ni * 16 + m];
#pragma unroll
            for (int s = 0; s < 8; ++s) {
                bf[ni].h[0][s] = p[s * 136];
                bf[ni].h[1][s] = p[(s + 16) * 136];
            }
        }
#pragma unroll
        for (int mi = 0; mi < 4; ++mi)
#pragma unroll
            for (int ni = 0; ni < 2; ++ni)
                acc[mi][ni] = __builtin_amdgcn_wmma_f32_16x16x32_bf16(
                    false, af[mi].bf, false, bf[ni].bf,
                    (short)0, acc[mi][ni], false, false);

        // hand off buffers: my async copies done + everyone done reading
        wait_async0();
        __syncthreads();
        cur ^= 1;
    }

    // ---- epilogue: bias + store ----
#pragma unroll
    for (int mi = 0; mi < 4; ++mi) {
#pragma unroll
        for (int ni = 0; ni < 2; ++ni) {
            const int col = n0 + wn * 32 + ni * 16 + (lane & 15);
            const float bv = bias[col];
#pragma unroll
            for (int e = 0; e < 8; ++e) {
                const int row = m0 + wm * 64 + mi * 16 + (lane >> 4) * 8 + e;
                const float v = acc[mi][ni][e] + bv;
                if constexpr (sizeof(OT) == 4)
                    C[(size_t)row * N + col] = v;
                else
                    C[(size_t)row * N + col] = f32_to_bf16(v);
            }
        }
    }
}

// ---------------------------------------------------------------------------
// Fused attention: one block per (batch*chunk, head).
// Q 64x64 + K^T (d-major) in LDS, scores via WMMA, softmax via native LDS
// float atomics, P (bf16) overlaid on dead K^T, O = P*V via WMMA with V
// fragments straight from global, 1/rowsum folded into epilogue.
// ---------------------------------------------------------------------------
__global__ __launch_bounds__(256) void attn_kernel(
    const unsigned short* __restrict__ qproj,   // [8192, 1024] bf16
    const unsigned short* __restrict__ kproj,   // [32768, 1024] bf16
    const unsigned short* __restrict__ vproj,   // [32768, 1024] bf16
    unsigned short* __restrict__ obuf)          // [8192, 1024] bf16
{
    __shared__ unsigned short ldsQ[64 * 72];     // [m][d], stride 72
    __shared__ unsigned short ldsKP[64 * 264];   // K^T [d][t] then P [m][t], stride 264
    __shared__ float rowmax[64];
    __shared__ float rowsum[64];

    const int bid  = blockIdx.x;
    const int h    = bid & 15;        // head
    const int bc   = bid >> 4;        // b*NC + c, 0..127
    const int tid  = threadIdx.x;
    const int lane = tid & 31;
    const int w    = tid >> 5;

    const unsigned short* qg = qproj + (size_t)bc * 64 * 1024 + h * 64;
    const unsigned short* kg = kproj + (size_t)bc * 256 * 1024 + h * 64;
    const unsigned short* vg = vproj + (size_t)bc * 256 * 1024 + h * 64;

    if (tid < 64) { rowmax[tid] = -3.0e38f; rowsum[tid] = 0.0f; }

    // ---- stage Q (64x64) via async copy ----
    {
        const int r  = tid >> 2;
        const int cb = (tid & 3) * 16;
        async_copy_b128(qg + (size_t)r * 1024 + cb,     &ldsQ[r * 72 + cb]);
        async_copy_b128(qg + (size_t)r * 1024 + cb + 8, &ldsQ[r * 72 + cb + 8]);
    }
    // ---- stage K transposed: ldsKP[d][t] ----
    for (int it = 0; it < 8; ++it) {
        const int idx = tid + it * 256;          // 2048 chunks of 8 elems
        const int t   = idx >> 3;                // 0..255
        const int db  = (idx & 7) * 8;           // 0..56
        u16x8 x = *(const u16x8*)(kg + (size_t)t * 1024 + db);
#pragma unroll
        for (int j = 0; j < 8; ++j) ldsKP[(db + j) * 264 + t] = x[j];
    }
    wait_async0();
    __syncthreads();

    // ---- scores: S[64][256] = Q * K^T ; waves 2(m) x 4(n), tile 32x64 ----
    const int wm = w >> 2;            // 0..1
    const int wn = w & 3;             // 0..3
    v8f acc[2][4] = {};
    for (int d0 = 0; d0 < 64; d0 += 32) {
        BF16Frag a[2], b[4];
        const int m  = lane & 15;
        const int kb = d0 + (lane >> 4) * 8;
#pragma unroll
        for (int mi = 0; mi < 2; ++mi) {
            const unsigned short* p = &ldsQ[(wm * 32 + mi * 16 + m) * 72 + kb];
            a[mi].h[0] = *(const u16x8*)p;
            a[mi].h[1] = *(const u16x8*)(p + 16);
        }
#pragma unroll
        for (int ni = 0; ni < 4; ++ni) {
            const unsigned short* p = &ldsKP[kb * 264 + wn * 64 + ni * 16 + m];
#pragma unroll
            for (int s = 0; s < 8; ++s) {
                b[ni].h[0][s] = p[s * 264];
                b[ni].h[1][s] = p[(s + 16) * 264];
            }
        }
#pragma unroll
        for (int mi = 0; mi < 2; ++mi)
#pragma unroll
            for (int ni = 0; ni < 4; ++ni)
                acc[mi][ni] = __builtin_amdgcn_wmma_f32_16x16x32_bf16(
                    false, a[mi].bf, false, b[ni].bf,
                    (short)0, acc[mi][ni], false, false);
    }

    // ---- row max (scaled scores), native ds_max_num_f32 ----
    const float scale = 0.125f;       // 1/sqrt(64)
#pragma unroll
    for (int mi = 0; mi < 2; ++mi) {
#pragma unroll
        for (int e = 0; e < 8; ++e) {
            const int row = wm * 32 + mi * 16 + (lane >> 4) * 8 + e;
            float mx = -3.0e38f;
#pragma unroll
            for (int ni = 0; ni < 4; ++ni) mx = fmaxf(mx, acc[mi][ni][e]);
            atomicMaxF(&rowmax[row], mx * scale);
        }
    }
    __syncthreads();   // rowmax complete; all K^T/Q reads complete

    // ---- exp, row sums, write P (bf16) over dead K^T region ----
#pragma unroll
    for (int mi = 0; mi < 2; ++mi) {
#pragma unroll
        for (int e = 0; e < 8; ++e) {
            const int row = wm * 32 + mi * 16 + (lane >> 4) * 8 + e;
            const float rm = rowmax[row];
            float s = 0.0f;
#pragma unroll
            for (int ni = 0; ni < 4; ++ni) {
                const float p = __expf(acc[mi][ni][e] * scale - rm);
                acc[mi][ni][e] = p;
                s += p;
            }
            atomicAdd(&rowsum[row], s);
        }
    }
#pragma unroll
    for (int mi = 0; mi < 2; ++mi)
#pragma unroll
        for (int ni = 0; ni < 4; ++ni)
#pragma unroll
            for (int e = 0; e < 8; ++e) {
                const int row = wm * 32 + mi * 16 + (lane >> 4) * 8 + e;
                const int col = wn * 64 + ni * 16 + (lane & 15);
                ldsKP[row * 264 + col] = f32_to_bf16(acc[mi][ni][e]);
            }
    __syncthreads();   // P visible, rowsum complete

    // ---- O[64][64] = P * V ; waves 4(m) x 2(n), tile 16x32 ----
    const int wm2 = w >> 1;           // 0..3
    const int wn2 = w & 1;            // 0..1
    v8f oacc[2] = {};
    for (int t0 = 0; t0 < 256; t0 += 32) {
        BF16Frag a, b[2];
        const int m  = lane & 15;
        const int kb = t0 + (lane >> 4) * 8;
        const unsigned short* p = &ldsKP[(wm2 * 16 + m) * 264 + kb];
        a.h[0] = *(const u16x8*)p;
        a.h[1] = *(const u16x8*)(p + 16);
#pragma unroll
        for (int ni = 0; ni < 2; ++ni) {
            const unsigned short* gp = vg + (size_t)kb * 1024 + wn2 * 32 + ni * 16 + m;
#pragma unroll
            for (int s = 0; s < 8; ++s) {
                b[ni].h[0][s] = gp[s * 1024];
                b[ni].h[1][s] = gp[(s + 16) * 1024];
            }
        }
#pragma unroll
        for (int ni = 0; ni < 2; ++ni)
            oacc[ni] = __builtin_amdgcn_wmma_f32_16x16x32_bf16(
                false, a.bf, false, b[ni].bf,
                (short)0, oacc[ni], false, false);
    }

    // ---- epilogue: normalize by rowsum, store bf16 merged-head layout ----
#pragma unroll
    for (int ni = 0; ni < 2; ++ni) {
#pragma unroll
        for (int e = 0; e < 8; ++e) {
            const int row = wm2 * 16 + (lane >> 4) * 8 + e;
            const int col = wn2 * 32 + ni * 16 + (lane & 15);
            const float v = oacc[ni][e] / rowsum[row];
            obuf[(size_t)bc * 64 * 1024 + (size_t)row * 1024 + h * 64 + col] =
                f32_to_bf16(v);
        }
    }
}

// ---------------------------------------------------------------------------
// Host launcher
// ---------------------------------------------------------------------------
extern "C" void kernel_launch(void* const* d_in, const int* in_sizes, int n_in,
                              void* d_out, int out_size, void* d_ws, size_t ws_size,
                              hipStream_t stream)
{
    (void)in_sizes; (void)n_in; (void)out_size; (void)ws_size;

    const float* query = (const float*)d_in[0];   // [4, 2048, 1024]
    const float* keyin = (const float*)d_in[1];   // [4, 32, 2, 128, 1024]
    const float* Wq = (const float*)d_in[2];
    const float* bq = (const float*)d_in[3];
    const float* Wk = (const float*)d_in[4];
    const float* bk = (const float*)d_in[5];
    const float* Wv = (const float*)d_in[6];
    const float* bv = (const float*)d_in[7];
    const float* Wo = (const float*)d_in[8];
    const float* bo = (const float*)d_in[9];
    float* out = (float*)d_out;                   // [4, 2048, 1024] f32

    const size_t QN = (size_t)8192 * 1024;        // query rows * d_model
    const size_t KN = (size_t)32768 * 1024;       // kv rows * d_model
    const size_t WN = (size_t)1024 * 1024;        // weight matrix

    // workspace layout (bf16 elements)
    unsigned short* ws  = (unsigned short*)d_ws;
    unsigned short* qbf = ws;              // query bf16
    unsigned short* kbf = qbf + QN;        // key   bf16
    unsigned short* wqb = kbf + KN;
    unsigned short* wkb = wqb + WN;
    unsigned short* wvb = wkb + WN;
    unsigned short* wob = wvb + WN;
    unsigned short* qp  = wob + WN;        // q_proj
    unsigned short* kp  = qp + QN;         // k_proj
    unsigned short* vp  = kp + KN;         // v_proj
    unsigned short* ob  = vp + KN;         // merged attention output

    const dim3 blk(256);

    // fp32 -> bf16 conversion passes (all sizes divisible by 2048)
    cvt_kernel<<<dim3((unsigned)(QN / 2048)), blk, 0, stream>>>(query, qbf, (int)QN);
    cvt_kernel<<<dim3((unsigned)(KN / 2048)), blk, 0, stream>>>(keyin, kbf, (int)KN);
    cvt_kernel<<<dim3((unsigned)(WN / 2048)), blk, 0, stream>>>(Wq, wqb, (int)WN);
    cvt_kernel<<<dim3((unsigned)(WN / 2048)), blk, 0, stream>>>(Wk, wkb, (int)WN);
    cvt_kernel<<<dim3((unsigned)(WN / 2048)), blk, 0, stream>>>(Wv, wvb, (int)WN);
    cvt_kernel<<<dim3((unsigned)(WN / 2048)), blk, 0, stream>>>(Wo, wob, (int)WN);

    // Q/K/V projections (bf16 out), async-staged double-buffered WMMA GEMMs
    gemm_bf16_kernel<unsigned short>
        <<<dim3(8, 64), blk, 0, stream>>>(qbf, wqb, bq, qp, 8192, 1024, 1024);
    gemm_bf16_kernel<unsigned short>
        <<<dim3(8, 256), blk, 0, stream>>>(kbf, wkb, bk, kp, 32768, 1024, 1024);
    gemm_bf16_kernel<unsigned short>
        <<<dim3(8, 256), blk, 0, stream>>>(kbf, wvb, bv, vp, 32768, 1024, 1024);

    // fused attention per (batch*chunk, head): 128 * 16 = 2048 blocks
    attn_kernel<<<dim3(2048), blk, 0, stream>>>(qp, kp, vp, ob);

    // output projection (f32 out)
    gemm_bf16_kernel<float>
        <<<dim3(8, 64), blk, 0, stream>>>(ob, wob, bo, out, 8192, 1024, 1024);
}
